// GCNGraph_14525579395558
// MI455X (gfx1250) — matrix operand order repaired and checked by reading, not compiled, not run on hardware
//
#include <hip/hip_runtime.h>
#include <hip/hip_bf16.h>
#include <stdint.h>

// ---------------- problem constants (match reference) ----------------
constexpr int N_NODES = 50000;
constexpr int N_EDGES = 800000;
constexpr int F_IN    = 128;
constexpr int HID     = 128;
constexpr int N_CLS   = 10;
constexpr int N_GRAPH = 512;

// ---------------- vector types ----------------
typedef __attribute__((ext_vector_type(16))) __bf16        v16bf;
typedef __attribute__((ext_vector_type(8)))  float         v8f;
typedef __attribute__((ext_vector_type(8)))  unsigned int  v8u;
typedef __attribute__((ext_vector_type(4)))  unsigned int  v4u;
typedef __attribute__((ext_vector_type(8)))  int           i32x8;
typedef __attribute__((ext_vector_type(4)))  int           i32x4;

#if defined(__has_builtin)
#if __has_builtin(__builtin_amdgcn_tensor_load_to_lds) && __has_builtin(__builtin_amdgcn_s_wait_tensorcnt)
#define GCN_USE_TDM 1
#endif
#endif

static __device__ __forceinline__ unsigned short f32_to_bf16(float f) {
    unsigned int u = __float_as_uint(f);
    u += 0x7fffu + ((u >> 16) & 1u);   // round-to-nearest-even
    return (unsigned short)(u >> 16);
}

static __device__ __forceinline__ void atomic_add_f32(float* p, float v) {
    // relaxed, device scope -> global_atomic_add_f32 on gfx1250
    __hip_atomic_fetch_add(p, v, __ATOMIC_RELAXED, __HIP_MEMORY_SCOPE_AGENT);
}

// ---------------- small elementwise kernels ----------------

__global__ void k_init_deg(float* deg) {
    int i = blockIdx.x * blockDim.x + threadIdx.x;
    if (i < N_NODES) deg[i] = 1.0f;   // +1 self loop
}

__global__ void k_deg_edges(const long long* __restrict__ ei, float* deg) {
    int e = blockIdx.x * blockDim.x + threadIdx.x;
    if (e < N_EDGES) {
        long long dst = ei[N_EDGES + e];
        atomic_add_f32(&deg[dst], 1.0f);
    }
}

__global__ void k_rsqrt_inplace(float* deg) {
    int i = blockIdx.x * blockDim.x + threadIdx.x;
    if (i < N_NODES) deg[i] = rsqrtf(deg[i]);
}

// cast fp32 row-major [N][128] -> packed bf16 pairs [N][64] (u32)
__global__ void k_cast_x(const float* __restrict__ x, unsigned int* __restrict__ xp) {
    int t = blockIdx.x * blockDim.x + threadIdx.x;
    if (t < N_NODES * 64) {
        float a = x[2 * t];
        float b = x[2 * t + 1];
        xp[t] = (unsigned int)f32_to_bf16(a) | ((unsigned int)f32_to_bf16(b) << 16);
    }
}

// pack weight W [128][128] row-major (k,n) -> WP[n][j] = pack(W[2j][n], W[2j+1][n])
__global__ void k_pack_w(const float* __restrict__ W, unsigned int* __restrict__ WP) {
    int t = blockIdx.x * blockDim.x + threadIdx.x;
    if (t < 128 * 64) {
        int n = t >> 6;
        int j = t & 63;
        float lo = W[(2 * j) * 128 + n];
        float hi = W[(2 * j + 1) * 128 + n];
        WP[t] = (unsigned int)f32_to_bf16(lo) | ((unsigned int)f32_to_bf16(hi) << 16);
    }
}

// ---------------- WMMA GEMM: Y[nrows][128] = Xp(bf16 pairs) @ W(packed) ----------------
// block = 256 threads = 8 waves; each wave -> 16 rows x 128 cols.
__global__ __launch_bounds__(256) void k_gemm_bf16(
    const unsigned int* __restrict__ Xp,   // [nrows][64] packed bf16 pairs, row-major
    const unsigned int* __restrict__ WP,   // [128][64]   packed, column(n)-major pairs
    float* __restrict__ Y,                 // [nrows][128]
    int nrows)
{
    __shared__ unsigned int sW[128 * 64];  // 32 KB

#ifdef GCN_USE_TDM
    // Stage the 32KB packed weight block with one Tensor-Data-Mover DMA (wave 0 only).
    if (threadIdx.x < 32) {
        unsigned int  lds_off = (unsigned int)(uint64_t)(size_t)&sW[0]; // low 32 bits = LDS offset
        unsigned long long ga = (unsigned long long)(size_t)WP;
        // D# group0: [1:0]=count=1, [63:32]=lds_addr, [120:64]=global_addr, [127:126]=type(2)
        v4u g0 = { 1u,
                   lds_off,
                   (unsigned int)(ga & 0xFFFFFFFFull),
                   (unsigned int)((ga >> 32) & 0x01FFFFFFull) | (2u << 30) };
        // D# group1: data_size=4B(code 2) @[17:16]; tensor_dim0=8192 @[79:48];
        //            tensor_dim1=1 @[111:80]; tile_dim0=8192 @[127:112]; tile_dim1=1 @[143:128];
        //            tensor_dim0_stride=8192 @[207:160]
        i32x8 g1 = { (int)(2u << 16),
                     (int)(8192u << 16),
                     (int)(1u << 16),
                     (int)(8192u << 16),
                     1,
                     8192,
                     0,
                     0 };
        i32x4 gz4 = { 0, 0, 0, 0 };
        i32x8 gz8 = { 0, 0, 0, 0, 0, 0, 0, 0 };
        // clang-23 / therock-10.0 lane: 6-arg form (g0, g1, g2, g3, g4, cpol)
        __builtin_amdgcn_tensor_load_to_lds(g0, g1, gz4, gz4, gz8, 0);
        __builtin_amdgcn_s_wait_tensorcnt(0);
    }
#else
    {
        const v4u* src = (const v4u*)WP;
        v4u* dst = (v4u*)sW;
        for (int i = threadIdx.x; i < 128 * 16; i += 256) dst[i] = src[i];
    }
#endif
    __syncthreads();

    const int wave = threadIdx.x >> 5;
    const int lane = threadIdx.x & 31;
    const int m    = lane & 15;    // row within A tile / col within B,C tile
    const int h    = lane >> 4;    // K-half selector

    const int rowBase = blockIdx.x * 128 + wave * 16;
    int r  = rowBase + m;
    int rc = (r < nrows) ? r : (nrows - 1);

    v8f acc[8] = {};

    const v4u* xrow = (const v4u*)(Xp + (size_t)rc * 64);

#pragma unroll
    for (int kt = 0; kt < 4; ++kt) {
        // A fragment: lane holds row rc, K = kt*32 + {h*8+0..7, 16+h*8+0..7}
        v4u a0 = xrow[kt * 4 + h];         // pairs j = kt*16 + h*4 .. +3
        v4u a1 = xrow[kt * 4 + h + 2];     // pairs j = kt*16 + h*4 + 8 ..
        v8u av = {a0.x, a0.y, a0.z, a0.w, a1.x, a1.y, a1.z, a1.w};
        v16bf afrag = __builtin_bit_cast(v16bf, av);

        // Preload ALL 8 B fragments of this K-step (one big ds_load clause),
        // then fire the 8 WMMAs back-to-back so LDS latency overlaps matrix math.
        v16bf bfr[8];
#pragma unroll
        for (int nt = 0; nt < 8; ++nt) {
            const v4u* wrow = (const v4u*)(sW + (nt * 16 + m) * 64 + kt * 16 + h * 4);
            v4u b0 = wrow[0];
            v4u b1 = wrow[2];
            v8u bv = {b0.x, b0.y, b0.z, b0.w, b1.x, b1.y, b1.z, b1.w};
            bfr[nt] = __builtin_bit_cast(v16bf, bv);
        }
#pragma unroll
        for (int nt = 0; nt < 8; ++nt) {
            acc[nt] = __builtin_amdgcn_wmma_f32_16x16x32_bf16(
                false, afrag, false, bfr[nt], (short)0, acc[nt], false, false);
        }
    }

    // D layout: VGPR i -> M = h*8 + i ; N = nt*16 + m
#pragma unroll
    for (int nt = 0; nt < 8; ++nt) {
#pragma unroll
        for (int i = 0; i < 8; ++i) {
            int row = rowBase + h * 8 + i;
            if (row < nrows) Y[(size_t)row * 128 + nt * 16 + m] = acc[nt][i];
        }
    }
}

// ---------------- aggregation ----------------

// agg[n][f] = h[n][f] * dis[n]^2   (self-loop term, full overwrite)
__global__ void k_selfloop(const float* __restrict__ h, const float* __restrict__ dis,
                           float* __restrict__ agg) {
    int t = blockIdx.x * blockDim.x + threadIdx.x;
    if (t < N_NODES * 128) {
        int n = t >> 7;
        float d = dis[n];
        agg[t] = h[t] * d * d;
    }
}

// per (edge, 4-feature chunk): agg[dst] += h[src] * dis[src]*dis[dst]
__global__ void k_edge_scatter(const long long* __restrict__ ei,
                               const float* __restrict__ h,
                               const float* __restrict__ dis,
                               float* __restrict__ agg) {
    int t = blockIdx.x * blockDim.x + threadIdx.x;
    if (t < N_EDGES * 32) {
        int e = t >> 5;
        int c = (t & 31) * 4;
        long long s = ei[e];
        long long d = ei[N_EDGES + e];
        float norm = dis[s] * dis[d];
        const float4 v = *(const float4*)(h + (size_t)s * 128 + c);
        float* ap = agg + (size_t)d * 128 + c;
        atomic_add_f32(ap + 0, v.x * norm);
        atomic_add_f32(ap + 1, v.y * norm);
        atomic_add_f32(ap + 2, v.z * norm);
        atomic_add_f32(ap + 3, v.w * norm);
    }
}

// out_f32[n][f] = relu(agg[n][f] + b[f]); also repack to bf16 pairs for next GEMM
__global__ void k_bias_relu_cast(const float* __restrict__ agg, const float* __restrict__ b,
                                 float* __restrict__ hf, unsigned int* __restrict__ hp) {
    int t = blockIdx.x * blockDim.x + threadIdx.x;
    if (t < N_NODES * 64) {
        int n = t >> 6;
        int f0 = (t & 63) * 2;
        float v0 = fmaxf(agg[(size_t)n * 128 + f0] + b[f0], 0.0f);
        float v1 = fmaxf(agg[(size_t)n * 128 + f0 + 1] + b[f0 + 1], 0.0f);
        hf[(size_t)n * 128 + f0]     = v0;
        hf[(size_t)n * 128 + f0 + 1] = v1;
        hp[t] = (unsigned int)f32_to_bf16(v0) | ((unsigned int)f32_to_bf16(v1) << 16);
    }
}

// ---------------- pooling + classifier ----------------

__global__ void k_zero_pool(float* __restrict__ sums, float* __restrict__ counts) {
    int t = blockIdx.x * blockDim.x + threadIdx.x;
    if (t < N_GRAPH * 128) sums[t] = 0.0f;
    if (t < N_GRAPH) counts[t] = 0.0f;
}

__global__ void k_pool_accum(const float* __restrict__ h, const long long* __restrict__ batch,
                             float* __restrict__ sums, float* __restrict__ counts) {
    int t = blockIdx.x * blockDim.x + threadIdx.x;
    if (t < N_NODES * 128) {
        int n = t >> 7;
        int f = t & 127;
        long long g = batch[n];
        atomic_add_f32(&sums[(size_t)g * 128 + f], h[t]);
        if (f == 0) atomic_add_f32(&counts[g], 1.0f);
    }
}

__global__ void k_classifier(const float* __restrict__ sums, const float* __restrict__ counts,
                             const float* __restrict__ Wl, const float* __restrict__ bl,
                             float* __restrict__ out) {
    int t = blockIdx.x * blockDim.x + threadIdx.x;
    if (t < N_GRAPH * N_CLS) {
        int g = t / N_CLS;
        int c = t % N_CLS;
        float inv = 1.0f / fmaxf(counts[g], 1.0f);
        float acc = bl[c];
        const float* srow = sums + (size_t)g * 128;
#pragma unroll 8
        for (int k = 0; k < 128; ++k)
            acc += srow[k] * inv * Wl[k * N_CLS + c];
        out[t] = acc;
    }
}

// ---------------- host launcher ----------------

extern "C" void kernel_launch(void* const* d_in, const int* in_sizes, int n_in,
                              void* d_out, int out_size, void* d_ws, size_t ws_size,
                              hipStream_t stream) {
    (void)in_sizes; (void)n_in; (void)out_size; (void)ws_size;

    const float*     x     = (const float*)d_in[0];
    const long long* ei    = (const long long*)d_in[1];
    const long long* batch = (const long long*)d_in[2];
    const float*     W1    = (const float*)d_in[3];
    const float*     b1    = (const float*)d_in[4];
    const float*     W2    = (const float*)d_in[5];
    const float*     b2    = (const float*)d_in[6];
    const float*     Wl    = (const float*)d_in[7];
    const float*     bl    = (const float*)d_in[8];
    float*           out   = (float*)d_out;

    // workspace carving (256B aligned)
    char* ws = (char*)d_ws;
    size_t off = 0;
    auto carve = [&](size_t bytes) -> char* {
        char* p = ws + off;
        off = (off + bytes + 255) & ~(size_t)255;
        return p;
    };
    float*        bufA   = (float*)carve((size_t)N_NODES * 128 * 4); // h_lin / agg2 / scratch
    float*        bufB   = (float*)carve((size_t)N_NODES * 128 * 4); // agg1 / h2_lin / h2
    unsigned int* bufX   = (unsigned int*)carve((size_t)N_NODES * 64 * 4); // packed bf16 acts
    float*        dis    = (float*)carve((size_t)N_NODES * 4);
    unsigned int* WP1    = (unsigned int*)carve(128 * 64 * 4);
    unsigned int* WP2    = (unsigned int*)carve(128 * 64 * 4);
    float*        sums   = (float*)carve((size_t)N_GRAPH * 128 * 4);
    float*        counts = (float*)carve((size_t)N_GRAPH * 4);

    const int T = 256;
    auto blocks = [](long long n, int t) { return (int)((n + t - 1) / t); };

    // degree / normalization
    k_init_deg<<<blocks(N_NODES, T), T, 0, stream>>>(dis);
    k_deg_edges<<<blocks(N_EDGES, T), T, 0, stream>>>(ei, dis);
    k_rsqrt_inplace<<<blocks(N_NODES, T), T, 0, stream>>>(dis);

    // casts / weight packing
    k_cast_x<<<blocks((long long)N_NODES * 64, T), T, 0, stream>>>(x, bufX);
    k_pack_w<<<blocks(128 * 64, T), T, 0, stream>>>(W1, WP1);
    k_pack_w<<<blocks(128 * 64, T), T, 0, stream>>>(W2, WP2);

    const int gemmBlocks = (N_NODES + 127) / 128;

    // ---- layer 1 ----
    k_gemm_bf16<<<gemmBlocks, 256, 0, stream>>>(bufX, WP1, bufA, N_NODES);   // h1_lin -> bufA
    k_selfloop<<<blocks((long long)N_NODES * 128, T), T, 0, stream>>>(bufA, dis, bufB);
    k_edge_scatter<<<blocks((long long)N_EDGES * 32, T), T, 0, stream>>>(ei, bufA, dis, bufB);
    k_bias_relu_cast<<<blocks((long long)N_NODES * 64, T), T, 0, stream>>>(bufB, b1, bufA, bufX);

    // ---- layer 2 ----
    k_gemm_bf16<<<gemmBlocks, 256, 0, stream>>>(bufX, WP2, bufB, N_NODES);   // h2_lin -> bufB
    k_selfloop<<<blocks((long long)N_NODES * 128, T), T, 0, stream>>>(bufB, dis, bufA);
    k_edge_scatter<<<blocks((long long)N_EDGES * 32, T), T, 0, stream>>>(ei, bufB, dis, bufA);
    k_bias_relu_cast<<<blocks((long long)N_NODES * 64, T), T, 0, stream>>>(bufA, b2, bufB, bufX);

    // ---- pooling + classifier ----
    k_zero_pool<<<blocks(N_GRAPH * 128, T), T, 0, stream>>>(sums, counts);
    k_pool_accum<<<blocks((long long)N_NODES * 128, T), T, 0, stream>>>(bufB, batch, sums, counts);
    k_classifier<<<blocks(N_GRAPH * N_CLS, T), T, 0, stream>>>(sums, counts, Wl, bl, out);
}
